// SplineDQN_62843961475096
// MI455X (gfx1250) — matrix operand order repaired and checked by reading, not compiled, not run on hardware
//
#include <hip/hip_runtime.h>

// ---------------------------------------------------------------------------
// Fully fused SplineDQN forward for MI455X (gfx1250, wave32, WMMA).
//
// One workgroup (256 threads = 8 wave32) owns 32 batch rows end-to-end:
//   stage input tile f32->bf16 in LDS
//   GEMM1 (K=128)  -> +bias -> LN+ReLU -> bf16 act tile (LDS)
//   GEMM2 (K=512)  -> +bias -> LN+ReLU -> bf16 act tile (LDS)
//   GEMM3 (K=512, N=608 = 599 WV + Wa + Wb + pad) -> sp tile (LDS)
//   spline: softmax cumsums (shfl scan), binary search, rational quadratic
//
// Roofline: ~83 GFLOP bf16-WMMA; unavoidable HBM = 33.5MB in + 52MB out
// (~3.7us at 23.3 TB/s). Activations never leave LDS; bf16 weights (~1.25MB)
// stay L2-resident. M-tile = 32 rows: each B-fragment (32B) feeds TWO wmmas
// (two 16-row M-groups), halving L2 weight traffic vs a 16-row tile
// (~2.2GB total L2 reads). LDS use: 78.8KB sp tile + 33.3KB act = 112KB of
// the 320KB WGP budget.
// ---------------------------------------------------------------------------

#define K_BINS   200
#define NIN      128
#define HID      512
#define NPAD     608          // 599 (WV) + 1 (Wa) + 1 (Wb) + 7 zero pad -> 38 tiles
#define ROWS     32           // batch rows per workgroup (2 x 16-row M-groups)
#define RPW      4            // rows per wave for LN / spline (32 rows / 8 waves)
#define SP_STRIDE 616         // f32 sp tile row stride (padded vs 64-bank LDS)
#define ACT_S    520          // bf16 act tile row stride (512 + 8 pad, 16B aligned)
#define SPT_BYTES (ROWS * SP_STRIDE * 4)          // 78848
#define ACT_BYTES (ROWS * ACT_S * 2)              // 33280
#define SMEM_BYTES (SPT_BYTES + ACT_BYTES)        // 112128  (< 320KB WGP LDS)

typedef __bf16 v8bf  __attribute__((ext_vector_type(8)));
typedef __bf16 v16bf __attribute__((ext_vector_type(16)));
typedef float  v8f   __attribute__((ext_vector_type(8)));

static __device__ __forceinline__ v16bf bf_cat(v8bf lo, v8bf hi) {
  return __builtin_shufflevector(lo, hi, 0,1,2,3,4,5,6,7,8,9,10,11,12,13,14,15);
}

// A fragment 16x32 bf16 from LDS act tile (ISA 16-bit A layout).
// Lane L: row m = L%16; lanes 0-15 hold K {kk*32+0..7, +16..23},
// lanes 16-31 hold K {kk*32+8..15, +24..31}.
static __device__ __forceinline__ v16bf load_a_frag(const __bf16* act, int lane, int kk) {
  const int m  = lane & 15;
  const int kb = kk * 32 + (lane >> 4) * 8;
  const __bf16* p = act + m * ACT_S + kb;
  v8bf lo = *(const v8bf*)(p);
  v8bf hi = *(const v8bf*)(p + 16);
  return bf_cat(lo, hi);
}

// B fragment 32x16 bf16 from global weights stored row-major [N][Kdim]
// (B = W^T, so W rows are B columns: two 16B loads per lane).
// Lane L: col n = n0 + L%16; K = kk*32 + (L/16)*8 + {0..7, 16..23}.
static __device__ __forceinline__ v16bf load_b_frag(const __bf16* W, int Kdim,
                                                    int n0, int lane, int kk) {
  const int n  = n0 + (lane & 15);
  const int kb = kk * 32 + (lane >> 4) * 8;
  const __bf16* p = W + n * Kdim + kb;
  v8bf lo = *(const v8bf*)(p);
  v8bf hi = *(const v8bf*)(p + 16);
  return bf_cat(lo, hi);
}

static __device__ __forceinline__ v8f wmma_bf16(v16bf a, v16bf b, v8f c) {
  // (neg_a, A, neg_b, B, c_mod, C, reuse_a, reuse_b)
  return __builtin_amdgcn_wmma_f32_16x16x32_bf16(false, a, false, b, (short)0, c,
                                                 false, false);
}

static __device__ __forceinline__ float wave_sum(float v) {
  for (int m = 16; m; m >>= 1) v += __shfl_xor(v, m, 32);
  return v;
}
static __device__ __forceinline__ float wave_max(float v) {
  for (int m = 16; m; m >>= 1) v = fmaxf(v, __shfl_xor(v, m, 32));
  return v;
}
static __device__ __forceinline__ float wave_excl_scan(float v, int lane) {
  float run = v;
  for (int off = 1; off < 32; off <<= 1) {
    float t = __shfl_up(run, off, 32);
    if (lane >= off) run += t;
  }
  return run - v;
}
static __device__ __forceinline__ float softplusf(float x) {
  return (x > 20.0f) ? x : log1pf(expf(x));
}

// GEMM: 32 rows x 512 cols. A = LDS act tile (two 16-row M-groups),
// B = global W [512][Kdim]. Each wave owns 4 N-tiles (wave*64..+63) and BOTH
// M-groups, so every B fragment is consumed by two wmmas.
static __device__ __forceinline__ void gemm512(const __bf16* act, const __bf16* W,
                                               int Kdim, const float* bias,
                                               float* spT, int lane, int wave) {
  const int kIters = Kdim / 32;
  const int n0 = wave * 64;
  v8f c[2][4] = {};
  for (int kk = 0; kk < kIters; ++kk) {
    const v16bf a0 = load_a_frag(act,                lane, kk);   // rows  0..15
    const v16bf a1 = load_a_frag(act + 16 * ACT_S,   lane, kk);   // rows 16..31
    for (int t = 0; t < 4; ++t) {
      const v16bf b = load_b_frag(W, Kdim, n0 + t * 16, lane, kk);
      c[0][t] = wmma_bf16(a0, b, c[0][t]);
      c[1][t] = wmma_bf16(a1, b, c[1][t]);
    }
  }
  // D layout: VGPR j -> M = g*16 + j + (lane/16)*8, N = n0 + t*16 + lane%16
  const int mAdd = (lane >> 4) * 8;
  for (int t = 0; t < 4; ++t) {
    const int n = n0 + t * 16 + (lane & 15);
    const float bv = bias[n];
    for (int g = 0; g < 2; ++g)
      for (int j = 0; j < 8; ++j)
        spT[(g * 16 + j + mAdd) * SP_STRIDE + n] = c[g][t][j] + bv;
  }
}

// LayerNorm + ReLU over 512 cols; RPW rows per wave; writes bf16 act tile.
static __device__ __forceinline__ void ln_relu(const float* spT, const float* g,
                                               const float* be, __bf16* act,
                                               int lane, int wave) {
  for (int r2 = 0; r2 < RPW; ++r2) {
    const int row = wave * RPW + r2;
    const float* x = spT + row * SP_STRIDE;
    const int c0 = lane * 16;
    float s = 0.0f;
    for (int j = 0; j < 16; ++j) s += x[c0 + j];
    const float mean = wave_sum(s) * (1.0f / 512.0f);
    float v = 0.0f;
    for (int j = 0; j < 16; ++j) { float d = x[c0 + j] - mean; v += d * d; }
    const float rs = rsqrtf(wave_sum(v) * (1.0f / 512.0f) + 1e-5f);
    for (int j = 0; j < 16; ++j) {
      const int cc = c0 + j;
      float y = (x[cc] - mean) * rs * g[cc] + be[cc];
      y = fmaxf(y, 0.0f);
      act[row * ACT_S + cc] = (__bf16)y;
    }
  }
}

__global__ __launch_bounds__(256)
void spline_fused(const float* __restrict__ inputs,
                  const float* __restrict__ b1, const float* __restrict__ g1,
                  const float* __restrict__ be1,
                  const float* __restrict__ b2, const float* __restrict__ g2,
                  const float* __restrict__ be2,
                  const __bf16* __restrict__ w1b, const __bf16* __restrict__ w2b,
                  const __bf16* __restrict__ wvb, const float* __restrict__ bvp,
                  float* __restrict__ out) {
  extern __shared__ char smem[];
  float*  spT = (float*)smem;                       // [32][SP_STRIDE] f32
  __bf16* act = (__bf16*)(smem + SPT_BYTES);        // [32][ACT_S] bf16

  const int tid  = threadIdx.x;
  const int lane = tid & 31;
  const int wave = tid >> 5;
  const int rowBase = blockIdx.x * ROWS;

  // ---- stage inputs tile (32 x 128) f32 -> bf16 into act (NT loads: one-shot)
  {
    const int idx = tid * 16;                // 4096 elems / 256 thr = 16 each
    const int r = idx >> 7, cc = idx & 127;
    const float* src = inputs + (rowBase + r) * NIN + cc;
    __bf16* dst = act + r * ACT_S + cc;
    for (int j = 0; j < 16; ++j) dst[j] = (__bf16)__builtin_nontemporal_load(src + j);
  }
  __syncthreads();

  // ---- layer 1
  gemm512(act, w1b, NIN, b1, spT, lane, wave);
  __syncthreads();
  ln_relu(spT, g1, be1, act, lane, wave);
  __syncthreads();

  // ---- layer 2
  gemm512(act, w2b, HID, b2, spT, lane, wave);
  __syncthreads();
  ln_relu(spT, g2, be2, act, lane, wave);
  __syncthreads();

  // ---- head GEMM: N = 608 (WV rows 0..598, Wa at 599, Wb at 600, pad 0)
  for (int nt = wave; nt < NPAD / 16; nt += 8) {
    const int n0 = nt * 16;
    v8f c0 = {}, c1 = {};
    for (int kk = 0; kk < HID / 32; ++kk) {
      const v16bf a0 = load_a_frag(act,              lane, kk);
      const v16bf a1 = load_a_frag(act + 16 * ACT_S, lane, kk);
      const v16bf b  = load_b_frag(wvb, HID, n0, lane, kk);
      c0 = wmma_bf16(a0, b, c0);
      c1 = wmma_bf16(a1, b, c1);
    }
    const int n = n0 + (lane & 15);
    const float bv = bvp[n];
    const int mAdd = (lane >> 4) * 8;
    for (int j = 0; j < 8; ++j) {
      spT[(j + mAdd) * SP_STRIDE + n]      = c0[j] + bv;
      spT[(16 + j + mAdd) * SP_STRIDE + n] = c1[j] + bv;
    }
  }
  __syncthreads();

  // ---- spline: RPW rows per wave; cw/ch scratch overlays the dead act tile
  const float INVK  = 1.0f / (float)K_BINS;
  const float COEF  = 1.0f - 0.001f * (float)K_BINS;   // 0.8
  const float EDGE  = logf(expf(0.999f) - 1.0f);
  float* cw = (float*)(smem + SPT_BYTES) + wave * 416; // 201 used, 208 stride
  float* ch = cw + 208;

  for (int r2 = 0; r2 < RPW; ++r2) {
    const int row = wave * RPW + r2;
    const float* sp = spT + row * SP_STRIDE;
    const float scale_a = expf(sp[599]);
    const float scale_b = sp[600];

    const int i0 = lane * 7;                       // contiguous chunk per lane
    int cnt = 200 - i0;
    cnt = cnt < 0 ? 0 : (cnt > 7 ? 7 : cnt);

    // softmax stats for W and H logits
    float mW = -1e30f, mH = -1e30f;
    for (int j = 0; j < cnt; ++j) {
      mW = fmaxf(mW, sp[i0 + j]);
      mH = fmaxf(mH, sp[200 + i0 + j]);
    }
    mW = wave_max(mW); mH = wave_max(mH);
    float sW = 0.0f, sH = 0.0f;
    for (int j = 0; j < cnt; ++j) {
      sW += expf(sp[i0 + j] - mW);
      sH += expf(sp[200 + i0 + j] - mH);
    }
    const float invW = 1.0f / wave_sum(sW);
    const float invH = 1.0f / wave_sum(sH);
    float pW = wave_excl_scan(sW, lane);
    float pH = wave_excl_scan(sH, lane);

    // cumsum of Wp = MIN_W*(i+1) + COEF*prefixExp/sum ; ch has affine applied
    for (int j = 0; j < cnt; ++j) {
      const int i = i0 + j;
      pW += expf(sp[i] - mW);
      pH += expf(sp[200 + i] - mH);
      cw[i + 1] = 0.001f * (float)(i + 1) + COEF * pW * invW;
      ch[i + 1] = scale_a * (0.001f * (float)(i + 1) + COEF * pH * invH) + scale_b;
    }
    if (lane == 0) { cw[0] = 0.0f; cw[200] = 1.0f; ch[0] = scale_b; }
    __syncthreads();

    // gather: searchsorted(right)-1 then rational-quadratic spline
    for (int j = 0; j < cnt; ++j) {
      const int i = i0 + j;
      const float tau = ((float)i + 0.5f) * INVK;
      int lo = 0, hi = 200;
      while (lo < hi) {
        const int mid = (lo + hi) >> 1;
        if (cw[mid] <= tau) lo = mid + 1; else hi = mid;
      }
      int b = lo - 1;
      b = b < 0 ? 0 : (b > 199 ? 199 : b);
      const float in_cw = cw[b], w = cw[b + 1] - in_cw;
      const float in_ch = ch[b], h = ch[b + 1] - in_ch;
      const float delta = h / w;
      const float d0 = (b == 0)   ? EDGE : (0.001f + softplusf(sp[399 + b]));
      const float d1 = (b == 199) ? EDGE : (0.001f + softplusf(sp[400 + b]));
      const float th = (tau - in_cw) / w;
      const float t1mt = th * (1.0f - th);
      const float numer = h * (delta * th * th + d0 * t1mt);
      const float denom = delta + (d0 + d1 - 2.0f * delta) * t1mt;
      __builtin_nontemporal_store(in_ch + numer / denom,
                                  out + (rowBase + row) * K_BINS + i);
    }
    __syncthreads();
  }
}

// ---------------------------------------------------------------------------
// Weight packing: f32 -> bf16, WV/Wa/Wb fused into one padded [608][512]
// matrix, biases bV/ba/bb into one padded [608] vector. Runs every launch
// (deterministic, ~0.64M elements, negligible).
// ---------------------------------------------------------------------------
__global__ __launch_bounds__(256)
void pack_weights(const float* __restrict__ W1, const float* __restrict__ W2,
                  const float* __restrict__ WV, const float* __restrict__ Wa,
                  const float* __restrict__ Wb, const float* __restrict__ bV,
                  const float* __restrict__ ba, const float* __restrict__ bb,
                  __bf16* __restrict__ w1b, __bf16* __restrict__ w2b,
                  __bf16* __restrict__ wvb, float* __restrict__ bvp) {
  int id = blockIdx.x * 256 + threadIdx.x;
  const int N1 = HID * NIN, N2 = HID * HID, N3 = NPAD * HID;
  if (id < N1) { w1b[id] = (__bf16)W1[id]; return; }
  id -= N1;
  if (id < N2) { w2b[id] = (__bf16)W2[id]; return; }
  id -= N2;
  if (id < N3) {
    const int n = id / HID, k = id - n * HID;
    float v = 0.0f;
    if (n < 599) v = WV[n * HID + k];
    else if (n == 599) v = Wa[k];
    else if (n == 600) v = Wb[k];
    wvb[id] = (__bf16)v;
    return;
  }
  id -= N3;
  if (id < NPAD) {
    float v = 0.0f;
    if (id < 599) v = bV[id];
    else if (id == 599) v = ba[0];
    else if (id == 600) v = bb[0];
    bvp[id] = v;
  }
}

extern "C" void kernel_launch(void* const* d_in, const int* in_sizes, int n_in,
                              void* d_out, int out_size, void* d_ws, size_t ws_size,
                              hipStream_t stream) {
  const float* inputs = (const float*)d_in[0];
  const float* W1  = (const float*)d_in[1];
  const float* b1  = (const float*)d_in[2];
  const float* g1  = (const float*)d_in[3];
  const float* be1 = (const float*)d_in[4];
  const float* W2  = (const float*)d_in[5];
  const float* b2  = (const float*)d_in[6];
  const float* g2  = (const float*)d_in[7];
  const float* be2 = (const float*)d_in[8];
  const float* WV  = (const float*)d_in[9];
  const float* bV  = (const float*)d_in[10];
  const float* Wa  = (const float*)d_in[11];
  const float* ba  = (const float*)d_in[12];
  const float* Wb  = (const float*)d_in[13];
  const float* bb  = (const float*)d_in[14];
  float* out = (float*)d_out;

  // workspace carve-up (all 256B-aligned offsets)
  char* ws = (char*)d_ws;
  __bf16* w1b = (__bf16*)(ws);                             // 512*128*2 = 131072
  __bf16* w2b = (__bf16*)(ws + 131072);                    // 512*512*2 = 524288
  __bf16* wvb = (__bf16*)(ws + 131072 + 524288);           // 608*512*2 = 622592
  float*  bvp = (float*)(ws + 131072 + 524288 + 622592);   // 608*4

  const int packTotal = HID * NIN + HID * HID + NPAD * HID + NPAD;
  pack_weights<<<(packTotal + 255) / 256, 256, 0, stream>>>(
      W1, W2, WV, Wa, Wb, bV, ba, bb, w1b, w2b, wvb, bvp);

  spline_fused<<<65536 / ROWS, 256, SMEM_BYTES, stream>>>(
      inputs, b1, g1, be1, b2, g2, be2, w1b, w2b, wvb, bvp, out);
}
